// KitbashMTREngine_85289460563994
// MI455X (gfx1250) — matrix operand order repaired and checked by reading, not compiled, not run on hardware
//
#include <hip/hip_runtime.h>
#include <hip/hip_bf16.h>

// ---------------------------------------------------------------------------
// Types for CDNA5 WMMA (wave32, v_wmma_f32_16x16x32_bf16)
// ---------------------------------------------------------------------------
typedef __bf16 bf16_t;
typedef __attribute__((ext_vector_type(16))) __bf16 v16bf;
typedef __attribute__((ext_vector_type(8)))  __bf16 v8bf;
typedef __attribute__((ext_vector_type(8)))  float  v8f;

// Problem constants
constexpr int   B_   = 4;
constexpr int   S_   = 2048;
constexpr int   D_   = 1024;
constexpr int   N_   = 12;
constexpr int   DS_  = 144;           // N*N
constexpr int   M_   = B_ * S_;       // 8192 token rows
constexpr int   KPAD_ = 160;          // DS padded up to multiple of 32 for WMMA K-loop
constexpr int   NLAYERS_ = 6;
constexpr float LR_ = 0.1f, BASE_DECAY_ = 0.05f, KAPPA_ = 1.0f;
constexpr float MTR_TH_ = 0.3f, DELTA_TH_ = 0.4f, CP_TH_ = 0.85f;

// d_out layout (floats, concatenated tuple outputs)
constexpr size_t OUT_OUTPUT = 0;                                 // (B,S,D)
constexpr size_t OUT_ERR    = OUT_OUTPUT + (size_t)M_ * D_;      // (B,S,1)
constexpr size_t OUT_REPS   = OUT_ERR + (size_t)M_;              // (6,B,S,D)
constexpr size_t OUT_SAL    = OUT_REPS + (size_t)NLAYERS_*M_*D_; // (6,B,S,1)
constexpr size_t OUT_DIS    = OUT_SAL + (size_t)NLAYERS_*M_;     // (B,S,1)

// Workspace layout (bytes)
constexpr size_t WS_SGATE = 0;                                   // 8192 f32
constexpr size_t WS_POS   = WS_SGATE + 32768;                    // 8192 f32
constexpr size_t WS_GATE  = WS_POS   + 32768;                    // 8192 f32
constexpr size_t WS_H     = WS_GATE  + 32768;                    // 8192x1024 bf16
constexpr size_t WS_COMB  = WS_H     + (size_t)M_*D_*2;          // 8192x2048 bf16
constexpr size_t WS_KW    = WS_COMB  + (size_t)M_*2*D_*2;        // 144x1024 bf16
constexpr size_t WS_VW    = WS_KW    + (size_t)DS_*D_*2;
constexpr size_t WS_K     = WS_VW    + (size_t)DS_*D_*2;         // 8192x144 f32
constexpr size_t WS_V     = WS_K     + (size_t)M_*DS_*4;
constexpr size_t WS_YS    = WS_V     + (size_t)M_*DS_*4;         // 8192x160 bf16
constexpr size_t WS_OUTW  = WS_YS    + (size_t)M_*KPAD_*2;       // 1024x160 bf16
constexpr size_t WS_PROJW = WS_OUTW  + (size_t)D_*KPAD_*2;       // 6x1024x2048 bf16

// ---------------------------------------------------------------------------
// Helpers
// ---------------------------------------------------------------------------
__device__ inline unsigned short f32_to_bf16(float f) {
    unsigned int u = __float_as_uint(f);
    unsigned int r = u + 0x7FFFu + ((u >> 16) & 1u);   // round-to-nearest-even
    return (unsigned short)(r >> 16);
}
__device__ inline float bf16_to_f32(unsigned short h) {
    return __uint_as_float(((unsigned int)h) << 16);
}
__device__ inline float waveReduceSum(float v) {
#pragma unroll
    for (int off = 16; off > 0; off >>= 1) v += __shfl_down(v, off, 32);
    return v; // valid in lane 0
}
__device__ inline float sigmoidf(float z) { return 1.0f / (1.0f + __expf(-z)); }

// ---------------------------------------------------------------------------
// WMMA fragment loaders (wave32 layouts per CDNA5 ISA 7.12.2)
//  A frag (16x32, row-major A[M x lda]): lane l (half = l>>4): row += (l&15);
//    elems 0..7  = A[row][k0 + 8*half .. +7]
//    elems 8..15 = A[row][k0 + 8*half + 16 .. +23]
//  B frag (32x16 = W^T, W row-major [N x ldw]): lane l: col += (l&15);
//    elems 0..7 = W[col][k0 + 8*half .. +7], elems 8..15 = W[col][k0 + 8*half + 16 .. +23]
// ---------------------------------------------------------------------------
__device__ inline v16bf load_afrag(const bf16_t* __restrict__ arow, int k0, int half) {
    v8bf lo = *(const v8bf*)(arow + k0 + half * 8);
    v8bf hi = *(const v8bf*)(arow + k0 + half * 8 + 16);
    v16bf a;
#pragma unroll
    for (int i = 0; i < 8; ++i) { a[i] = lo[i]; a[i + 8] = hi[i]; }
    return a;
}
__device__ inline v16bf load_bfrag(const bf16_t* __restrict__ wrow, int k0, int half) {
    v8bf lo = *(const v8bf*)(wrow + k0 + half * 8);
    v8bf hi = *(const v8bf*)(wrow + k0 + half * 8 + 16);
    v16bf b;
#pragma unroll
    for (int i = 0; i < 8; ++i) { b[i] = lo[i]; b[i + 8] = hi[i]; }
    return b;
}
__device__ inline v8f wmma_bf16(v16bf a, v16bf b, v8f c) {
    return __builtin_amdgcn_wmma_f32_16x16x32_bf16(false, a, false, b,
                                                   (short)0, c, false, false);
}

// Pipelined 1x1 tile accumulate (16x16 out), used by the K/V projection.
__device__ inline v8f wmma_accum_1x1(const bf16_t* __restrict__ arow,
                                     const bf16_t* __restrict__ wrow,
                                     int K, int half) {
    v8f acc = {};
    v16bf a = load_afrag(arow, 0, half);
    v16bf b = load_bfrag(wrow, 0, half);
    for (int k0 = 32; k0 < K; k0 += 32) {
        v16bf an = load_afrag(arow, k0, half);   // next-step loads issued
        v16bf bn = load_bfrag(wrow, k0, half);   // before consuming current
        acc = wmma_bf16(a, b, acc);
        a = an; b = bn;
    }
    return wmma_bf16(a, b, acc);
}

// Pipelined 2x2 tile accumulate (32x32 out per wave): 4 WMMAs per 4 frag loads,
// A-fragments reused across both n-halves, software-pipelined double buffer.
__device__ inline void wmma_accum_2x2(const bf16_t* __restrict__ ar0,
                                      const bf16_t* __restrict__ ar1,
                                      const bf16_t* __restrict__ wr0,
                                      const bf16_t* __restrict__ wr1,
                                      int K, int half,
                                      v8f& c00, v8f& c01, v8f& c10, v8f& c11) {
    v16bf a0 = load_afrag(ar0, 0, half), a1 = load_afrag(ar1, 0, half);
    v16bf b0 = load_bfrag(wr0, 0, half), b1 = load_bfrag(wr1, 0, half);
    for (int k0 = 32; k0 < K; k0 += 32) {
        v16bf a0n = load_afrag(ar0, k0, half), a1n = load_afrag(ar1, k0, half);
        v16bf b0n = load_bfrag(wr0, k0, half), b1n = load_bfrag(wr1, k0, half);
        c00 = wmma_bf16(a0, b0, c00);
        c01 = wmma_bf16(a0, b1, c01);
        c10 = wmma_bf16(a1, b0, c10);
        c11 = wmma_bf16(a1, b1, c11);
        a0 = a0n; a1 = a1n; b0 = b0n; b1 = b1n;
    }
    c00 = wmma_bf16(a0, b0, c00);
    c01 = wmma_bf16(a0, b1, c01);
    c10 = wmma_bf16(a1, b0, c10);
    c11 = wmma_bf16(a1, b1, c11);
}

// ---------------------------------------------------------------------------
// Weight conversion kernels
// ---------------------------------------------------------------------------
__global__ void conv_bf16_kernel(const float* __restrict__ src,
                                 unsigned short* __restrict__ dst, int n) {
    int i = blockIdx.x * blockDim.x + threadIdx.x;
    if (i < n) dst[i] = f32_to_bf16(src[i]);
}

__global__ void conv_outw_pad_kernel(const float* __restrict__ out_w,
                                     unsigned short* __restrict__ dst) {
    int i = blockIdx.x * blockDim.x + threadIdx.x;      // over D_*KPAD_
    if (i >= D_ * KPAD_) return;
    int n = i / KPAD_, k = i % KPAD_;
    dst[i] = (k < DS_) ? f32_to_bf16(out_w[n * DS_ + k]) : (unsigned short)0;
}

// ---------------------------------------------------------------------------
// Stage A: copent gate, cumsum, positional embedding, h = x + pos_emb
// ---------------------------------------------------------------------------
__global__ void cp_gate_kernel(const float* __restrict__ x,
                               const float* __restrict__ cp_w,
                               const float* __restrict__ cp_b,
                               float* __restrict__ sgate) {
    int lane = threadIdx.x & 31;
    int m = blockIdx.x * (blockDim.x >> 5) + (threadIdx.x >> 5);
    if (m >= M_) return;
    const float* xr = x + (size_t)m * D_;
    float s = 0.f;
    for (int i = lane; i < D_; i += 32) s += xr[i] * cp_w[i];
    s = waveReduceSum(s);
    if (lane == 0) {
        float raw = sigmoidf(s + cp_b[0]);
        sgate[m] = (raw > CP_TH_) ? raw : 0.0f;
    }
}

__global__ void cumsum_kernel(const float* __restrict__ sgate,
                              float* __restrict__ pos) {
    __shared__ float psum[256];
    int b = blockIdx.x, tid = threadIdx.x;
    const float* src = sgate + (size_t)b * S_;
    int base = tid * 8;
    float loc[8];
    float run = 0.f;
#pragma unroll
    for (int i = 0; i < 8; ++i) { run += src[base + i]; loc[i] = run; }
    psum[tid] = run;
    __syncthreads();
    for (int off = 1; off < 256; off <<= 1) {
        float t = (tid >= off) ? psum[tid - off] : 0.f;
        __syncthreads();
        psum[tid] += t;
        __syncthreads();
    }
    float ofs = (tid > 0) ? psum[tid - 1] : 0.f;
#pragma unroll
    for (int i = 0; i < 8; ++i) pos[(size_t)b * S_ + base + i] = loc[i] + ofs;
}

__global__ void posemb_kernel(const float* __restrict__ x,
                              const float* __restrict__ pos,
                              unsigned short* __restrict__ h,
                              unsigned short* __restrict__ comb) {
    int row = blockIdx.x;        // token 0..8191
    int j = threadIdx.x;         // 0..511 frequency index
    float p = pos[row];
    // inv_freq = 10000^-(j/512)
    float invf = __expf(-(float)j * (9.210340371976184f / 512.0f));
    float ang = p * invf;
    float sv = __sinf(ang), cv = __cosf(ang);
    const float* xr = x + (size_t)row * D_;
    unsigned short* hr = h + (size_t)row * D_;
    unsigned short* cr = comb + (size_t)row * (2 * D_) + D_;  // second half
    hr[j]        = f32_to_bf16(xr[j] + sv);
    hr[j + 512]  = f32_to_bf16(xr[j + 512] + cv);
    cr[j]        = f32_to_bf16(sv);
    cr[j + 512]  = f32_to_bf16(cv);
}

// ---------------------------------------------------------------------------
// WMMA GEMM: K/V projections  (M=8192, N=144 -> 9 n-tiles, K=1024)
// ---------------------------------------------------------------------------
__global__ void gemm_kv_kernel(const unsigned short* __restrict__ A,
                               const unsigned short* __restrict__ W,
                               float* __restrict__ C,
                               int nTiles, int Kdim, int lda, int ldw, int ldc) {
    int lane = threadIdx.x;                       // 0..31
    int nt = blockIdx.y * 8 + threadIdx.y;
    if (nt >= nTiles) return;                     // uniform per wave
    int m0 = blockIdx.x * 16, n0 = nt * 16;
    int half = lane >> 4, l15 = lane & 15;
    const bf16_t* arow = (const bf16_t*)A + (size_t)(m0 + l15) * lda;
    const bf16_t* wrow = (const bf16_t*)W + (size_t)(n0 + l15) * ldw;
    v8f acc = wmma_accum_1x1(arow, wrow, Kdim, half);
    int col = n0 + l15;
#pragma unroll
    for (int r = 0; r < 8; ++r)
        C[(size_t)(m0 + 8 * half + r) * ldc + col] = acc[r];
}

// ---------------------------------------------------------------------------
// sel gate: gate = sigmoid(h . sel_w + sel_b)
// ---------------------------------------------------------------------------
__global__ void selgate_kernel(const unsigned short* __restrict__ h,
                               const float* __restrict__ sel_w,
                               const float* __restrict__ sel_b,
                               float* __restrict__ gate) {
    int lane = threadIdx.x & 31;
    int m = blockIdx.x * (blockDim.x >> 5) + (threadIdx.x >> 5);
    if (m >= M_) return;
    const unsigned short* hr = h + (size_t)m * D_;
    float s = 0.f;
    for (int i = lane; i < D_; i += 32) s += bf16_to_f32(hr[i]) * sel_w[i];
    s = waveReduceSum(s);
    if (lane == 0) gate[m] = sigmoidf(s + sel_b[0]);
}

// ---------------------------------------------------------------------------
// Sequential fast-weight scan: one block per batch, fast weights in LDS.
// ---------------------------------------------------------------------------
__global__ void scan_kernel(const float* __restrict__ Kin,
                            const float* __restrict__ Vin,
                            const float* __restrict__ gate,
                            unsigned short* __restrict__ ys,    // [M_][KPAD_] bf16
                            float* __restrict__ err_out) {      // d_out + OUT_ERR
    __shared__ float w1[N_ * N_ * N_], w2[N_ * N_ * N_];        // 1728 each
    __shared__ float kt[DS_], vt[DS_], hid[DS_], err[DS_], gh[DS_];
    __shared__ float act[N_], dlt[N_], decay[N_], strength[N_], last_seen[N_];
    __shared__ float timeS, sg, reconS;

    int b = blockIdx.x, tid = threadIdx.x;
    // init: w1 = w2 = broadcast(eye(N)) over i
    for (int idx = tid; idx < N_ * N_ * N_; idx += blockDim.x) {
        int jj = (idx / N_) % N_, kk = idx % N_;
        float e = (jj == kk) ? 1.0f : 0.0f;
        w1[idx] = e; w2[idx] = e;
    }
    if (tid < N_) { strength[tid] = 0.f; last_seen[tid] = 0.f; }
    if (tid == 0) timeS = 0.f;
    __syncthreads();

    for (int t = 0; t < S_; ++t) {
        size_t row = (size_t)b * S_ + t;
        if (tid < DS_) {
            kt[tid] = Kin[row * DS_ + tid];
            vt[tid] = Vin[row * DS_ + tid];
        }
        if (tid == 0) { sg = gate[row]; timeS += 1.0f; }
        __syncthreads();

        // hidden[i,k] = sum_j kt[i,j] * w1[i,j,k]; activity/decay per i
        if (tid < DS_) {
            int i = tid / N_, k = tid % N_;
            float s = 0.f;
#pragma unroll
            for (int j = 0; j < N_; ++j) s += kt[i * N_ + j] * w1[(i * N_ + j) * N_ + k];
            hid[tid] = s;
        }
        if (tid < N_) {
            float s2 = 0.f;
#pragma unroll
            for (int j = 0; j < N_; ++j) { float v = kt[tid * N_ + j]; s2 += v * v; }
            act[tid] = sqrtf(s2);
            float dt = timeS - last_seen[tid];
            dlt[tid] = dt;
            float tg = 1.0f / fmaxf(log1pf(dt * 0.1f), 1.0f);
            decay[tid] = BASE_DECAY_ / (1.0f + strength[tid] * tg);  // pre-update strength
        }
        __syncthreads();

        // vt_pred[i,k] = sum_j hid[j,i] * w2[i,j,k]; error; yt store (transposed)
        if (tid < DS_) {
            int i = tid / N_, k = tid % N_;
            float s = 0.f;
#pragma unroll
            for (int j = 0; j < N_; ++j) s += hid[j * N_ + i] * w2[(i * N_ + j) * N_ + k];
            err[tid] = s - vt[tid];
            ys[row * KPAD_ + (size_t)(k * N_ + i)] = f32_to_bf16(s);  // swapaxes(-1,-2)
        } else if (tid < KPAD_) {
            ys[row * KPAD_ + tid] = 0;                                // zero K-padding
        }
        __syncthreads();

        if (tid == 0) {
            float s = 0.f;
            for (int q = 0; q < DS_; ++q) s += err[q] * err[q];
            float en = sqrtf(s);
            err_out[row] = en;
            reconS = 1.0f / (1.0f + en);
        }
        // grad_h[i,j] = sum_k err[i,k] * w2[i,j,k]  (pre-update w2)
        if (tid < DS_) {
            int i = tid / N_, j = tid % N_;
            float s = 0.f;
#pragma unroll
            for (int k = 0; k < N_; ++k) s += err[i * N_ + k] * w2[(i * N_ + j) * N_ + k];
            gh[tid] = s;
        }
        __syncthreads();

        // weight update + decay toward eye
        float step = LR_ * sg;
        for (int idx = tid; idx < N_ * N_ * N_; idx += blockDim.x) {
            int i = idx / (N_ * N_), j = (idx / N_) % N_, k = idx % N_;
            float g1 = kt[i * N_ + j] * gh[i * N_ + k];
            float g2 = hid[j * N_ + i] * err[i * N_ + k];
            float df = decay[i];
            float eye = (j == k) ? 1.0f : 0.0f;
            w1[idx] = (w1[idx] - step * g1) * (1.0f - df) + eye * df;
            w2[idx] = (w2[idx] - step * g2) * (1.0f - df) + eye * df;
        }
        if (tid < N_) {
            strength[tid] += reconS * log1pf(dlt[tid]) * act[tid] * sg * 0.05f;
            if (act[tid] > 0.1f) last_seen[tid] = timeS;
        }
        __syncthreads();
    }
}

// ---------------------------------------------------------------------------
// WMMA GEMM (2x2 tiles): out = ys @ out_w^T + out_b
// writes output f32 (non-temporal, streamed) + combined bf16 (reused later)
// ---------------------------------------------------------------------------
__global__ void gemm_out_kernel(const unsigned short* __restrict__ A,   // ys
                                const unsigned short* __restrict__ W,   // outw pad
                                const float* __restrict__ out_b,
                                float* __restrict__ out,                // d_out
                                unsigned short* __restrict__ comb) {
    int lane = threadIdx.x, half = lane >> 4, l15 = lane & 15;
    int m0 = blockIdx.x * 32;
    int n0 = (blockIdx.y * 8 + threadIdx.y) * 32;
    const bf16_t* ar0 = (const bf16_t*)A + (size_t)(m0 + l15) * KPAD_;
    const bf16_t* ar1 = (const bf16_t*)A + (size_t)(m0 + 16 + l15) * KPAD_;
    const bf16_t* wr0 = (const bf16_t*)W + (size_t)(n0 + l15) * KPAD_;
    const bf16_t* wr1 = (const bf16_t*)W + (size_t)(n0 + 16 + l15) * KPAD_;
    v8f c00 = {}, c01 = {}, c10 = {}, c11 = {};
    wmma_accum_2x2(ar0, ar1, wr0, wr1, KPAD_, half, c00, c01, c10, c11);
#pragma unroll
    for (int p = 0; p < 2; ++p) {
        int rowb = m0 + 16 * p + 8 * half;
#pragma unroll
        for (int q = 0; q < 2; ++q) {
            int col = n0 + 16 * q + l15;
            float bias = out_b[col];
            const v8f& c = p ? (q ? c11 : c10) : (q ? c01 : c00);
#pragma unroll
            for (int r = 0; r < 8; ++r) {
                int rowi = rowb + r;
                float v = c[r] + bias;
                __builtin_nontemporal_store(v, &out[OUT_OUTPUT + (size_t)rowi * D_ + col]);
                comb[(size_t)rowi * (2 * D_) + col] = f32_to_bf16(v);
            }
        }
    }
}

// ---------------------------------------------------------------------------
// WMMA GEMM (2x2 tiles): reps[l] = relu(combined @ proj_w[l]^T + proj_b[l])
// 201 MB streamed out -> non-temporal stores (keep L2 for GEMM operands)
// ---------------------------------------------------------------------------
__global__ void gemm_reps_kernel(const unsigned short* __restrict__ A,   // combined
                                 const unsigned short* __restrict__ PW,  // projw bf16
                                 const float* __restrict__ proj_b,
                                 float* __restrict__ reps) {             // d_out+OUT_REPS
    int lane = threadIdx.x, half = lane >> 4, l15 = lane & 15;
    int l = blockIdx.z;
    int m0 = blockIdx.x * 32;
    int n0 = (blockIdx.y * 8 + threadIdx.y) * 32;
    const bf16_t* Ab = (const bf16_t*)A;
    const bf16_t* Wb = (const bf16_t*)PW + (size_t)l * D_ * (2 * D_);
    const bf16_t* ar0 = Ab + (size_t)(m0 + l15) * (2 * D_);
    const bf16_t* ar1 = Ab + (size_t)(m0 + 16 + l15) * (2 * D_);
    const bf16_t* wr0 = Wb + (size_t)(n0 + l15) * (2 * D_);
    const bf16_t* wr1 = Wb + (size_t)(n0 + 16 + l15) * (2 * D_);
    v8f c00 = {}, c01 = {}, c10 = {}, c11 = {};
    wmma_accum_2x2(ar0, ar1, wr0, wr1, 2 * D_, half, c00, c01, c10, c11);
    float* dst = reps + (size_t)l * M_ * D_;
#pragma unroll
    for (int p = 0; p < 2; ++p) {
        int rowb = m0 + 16 * p + 8 * half;
#pragma unroll
        for (int q = 0; q < 2; ++q) {
            int col = n0 + 16 * q + l15;
            float bias = proj_b[(size_t)l * D_ + col];
            const v8f& c = p ? (q ? c11 : c10) : (q ? c01 : c00);
#pragma unroll
            for (int r = 0; r < 8; ++r) {
                float v = fmaxf(c[r] + bias, 0.0f);
                __builtin_nontemporal_store(v, &dst[(size_t)(rowb + r) * D_ + col]);
            }
        }
    }
}

// ---------------------------------------------------------------------------
// sal[l, token] = sigmoid(KAPPA * (combined . sal_w[l] + sal_b[l]))
// ---------------------------------------------------------------------------
__global__ void sal_kernel(const unsigned short* __restrict__ comb,
                           const float* __restrict__ sal_w,
                           const float* __restrict__ sal_b,
                           float* __restrict__ out_sal) {
    int lane = threadIdx.x & 31;
    int wid = blockIdx.x * (blockDim.x >> 5) + (threadIdx.x >> 5);
    if (wid >= NLAYERS_ * M_) return;
    int l = wid / M_, m = wid % M_;
    const unsigned short* cr = comb + (size_t)m * (2 * D_);
    const float* wr = sal_w + (size_t)l * (2 * D_);
    float s = 0.f;
    for (int i = lane; i < 2 * D_; i += 32) s += bf16_to_f32(cr[i]) * wr[i];
    s = waveReduceSum(s);
    if (lane == 0)
        out_sal[(size_t)l * M_ + m] = sigmoidf(KAPPA_ * (s + sal_b[l]));
}

// ---------------------------------------------------------------------------
// dissonance = (err > MTR_TH) | (|sal0-sal2| > TH) | (|sal2-sal4| > TH)
// ---------------------------------------------------------------------------
__global__ void dissonance_kernel(float* __restrict__ out) {
    int t = blockIdx.x * blockDim.x + threadIdx.x;
    if (t >= M_) return;
    float er = out[OUT_ERR + t];
    float s0 = out[OUT_SAL + 0 * (size_t)M_ + t];
    float s2 = out[OUT_SAL + 2 * (size_t)M_ + t];
    float s4 = out[OUT_SAL + 4 * (size_t)M_ + t];
    bool d = (er > MTR_TH_) || (fabsf(s0 - s2) > DELTA_TH_) || (fabsf(s2 - s4) > DELTA_TH_);
    out[OUT_DIS + t] = d ? 1.0f : 0.0f;
}

// ---------------------------------------------------------------------------
// Host side
// ---------------------------------------------------------------------------
extern "C" void kernel_launch(void* const* d_in, const int* in_sizes, int n_in,
                              void* d_out, int out_size, void* d_ws, size_t ws_size,
                              hipStream_t stream) {
    const float* x      = (const float*)d_in[0];
    const float* k_w    = (const float*)d_in[1];
    const float* v_w    = (const float*)d_in[2];
    const float* sel_w  = (const float*)d_in[3];
    const float* sel_b  = (const float*)d_in[4];
    const float* out_w  = (const float*)d_in[5];
    const float* out_b  = (const float*)d_in[6];
    const float* cp_w   = (const float*)d_in[7];
    const float* cp_b   = (const float*)d_in[8];
    const float* proj_w = (const float*)d_in[9];
    const float* proj_b = (const float*)d_in[10];
    const float* sal_w  = (const float*)d_in[11];
    const float* sal_b  = (const float*)d_in[12];
    float* out = (float*)d_out;

    char* ws = (char*)d_ws;
    float*          sgate  = (float*)(ws + WS_SGATE);
    float*          pos    = (float*)(ws + WS_POS);
    float*          gate   = (float*)(ws + WS_GATE);
    unsigned short* h_bf   = (unsigned short*)(ws + WS_H);
    unsigned short* comb   = (unsigned short*)(ws + WS_COMB);
    unsigned short* kw_bf  = (unsigned short*)(ws + WS_KW);
    unsigned short* vw_bf  = (unsigned short*)(ws + WS_VW);
    float*          Kf     = (float*)(ws + WS_K);
    float*          Vf     = (float*)(ws + WS_V);
    unsigned short* ys_bf  = (unsigned short*)(ws + WS_YS);
    unsigned short* outw_bf= (unsigned short*)(ws + WS_OUTW);
    unsigned short* projw_bf=(unsigned short*)(ws + WS_PROJW);

    // --- weight conversions to bf16 ---
    {
        int n = DS_ * D_;
        conv_bf16_kernel<<<(n + 255) / 256, 256, 0, stream>>>(k_w, kw_bf, n);
        conv_bf16_kernel<<<(n + 255) / 256, 256, 0, stream>>>(v_w, vw_bf, n);
    }
    conv_outw_pad_kernel<<<(D_ * KPAD_ + 255) / 256, 256, 0, stream>>>(out_w, outw_bf);
    {
        int n = NLAYERS_ * D_ * 2 * D_;
        conv_bf16_kernel<<<(n + 255) / 256, 256, 0, stream>>>(proj_w, projw_bf, n);
    }

    // --- stage A: copent gate -> cumsum -> pos emb / h / combined[:,D:] ---
    cp_gate_kernel<<<M_ / 8, 256, 0, stream>>>(x, cp_w, cp_b, sgate);
    cumsum_kernel<<<B_, 256, 0, stream>>>(sgate, pos);
    posemb_kernel<<<M_, 512, 0, stream>>>(x, pos, h_bf, comb);

    // --- K / V projections via WMMA (M=8192, N=144->9 tiles, K=1024) ---
    {
        dim3 grid(M_ / 16, 2), block(32, 8);
        gemm_kv_kernel<<<grid, block, 0, stream>>>(h_bf, kw_bf, Kf, 9, D_, D_, D_, DS_);
        gemm_kv_kernel<<<grid, block, 0, stream>>>(h_bf, vw_bf, Vf, 9, D_, D_, D_, DS_);
    }
    selgate_kernel<<<M_ / 8, 256, 0, stream>>>(h_bf, sel_w, sel_b, gate);

    // --- sequential fast-weight scan (one block per batch, weights in LDS) ---
    scan_kernel<<<B_, 256, 0, stream>>>(Kf, Vf, gate, ys_bf, out + OUT_ERR);

    // --- output projection via WMMA 2x2 (K padded 144->160); fills combined[:,:D] ---
    {
        dim3 grid(M_ / 32, 4), block(32, 8);   // 32 n-tiles of 32
        gemm_out_kernel<<<grid, block, 0, stream>>>(ys_bf, outw_bf, out_b, out, comb);
    }

    // --- reps: 6 layers of (8192x2048)x(1024x2048)^T with bias + ReLU ---
    {
        dim3 grid(M_ / 32, 4, NLAYERS_), block(32, 8);
        gemm_reps_kernel<<<grid, block, 0, stream>>>(comb, projw_bf, proj_b,
                                                     out + OUT_REPS);
    }

    // --- sal + dissonance ---
    sal_kernel<<<(NLAYERS_ * M_) / 8, 256, 0, stream>>>(comb, sal_w, sal_b, out + OUT_SAL);
    dissonance_kernel<<<(M_ + 255) / 256, 256, 0, stream>>>(out);
}
